// ChroniqueLSTM_50534585205278
// MI455X (gfx1250) — matrix operand order repaired and checked by reading, not compiled, not run on hardware
//
#include <hip/hip_runtime.h>
#include <hip/hip_bf16.h>

typedef _Float16 v16h __attribute__((ext_vector_type(16)));
typedef _Float16 v8h  __attribute__((ext_vector_type(8)));
typedef float    v8f  __attribute__((ext_vector_type(8)));

#define BB   64
#define TT   1024
#define DD   256
#define HH   128
#define BT   (BB * TT)      // 65536
#define NG   512            // 4*H gate width
#define NTAG 3

// ---------------------------------------------------------------------------
// Fragment load: two contiguous 16B chunks -> one v16h operand.
// A-frag (16x32 f16): lane L holds row (L&15); elements 0..7 are K=kx..kx+7,
//   elements 8..15 are K=kx+16..kx+23 (kx = 8*(L>=16)).
// B-frag (32x16 f16): lane L holds col (L&15); same K order as the A-frag
//   (elements 0..7 K=kx..kx+7, 8..15 K=kx+16..kx+23).
__device__ __forceinline__ v16h ld_frag2(const _Float16* p0, const _Float16* p1) {
  const v8h lo = *(const v8h*)p0;
  const v8h hi = *(const v8h*)p1;
  return __builtin_shufflevector(lo, hi, 0, 1, 2, 3, 4, 5, 6, 7,
                                 8, 9, 10, 11, 12, 13, 14, 15);
}

__device__ __forceinline__ float sigmf(float x) {
  return 1.0f / (1.0f + __expf(-x));
}

// ---------------------------------------------------------------------------
// f32 -> f16 elementwise convert
__global__ void cvt_f16_kernel(const float* __restrict__ in, _Float16* __restrict__ out, int n) {
  for (int i = blockIdx.x * blockDim.x + threadIdx.x; i < n; i += gridDim.x * blockDim.x)
    out[i] = (_Float16)in[i];
}

// ---------------------------------------------------------------------------
// gx[t][b][n] = X(f16)[bt,256] @ W16^T + (b_ih + b_hh)
// W16 is [512,256] f16 row-major (fragment-contiguous). No LDS, no barriers;
// fragments stream from global (L2-hot) with 1-deep software pipelining so each
// k-chunk's loads are in flight behind the previous chunk's 4 WMMAs.
// Block: 256 thr (8 waves) -> 64(M) x 128(N) C tile. grid = (BT/64, 512/128).
__global__ __launch_bounds__(256) void gemm_gx_kernel(
    const _Float16* __restrict__ x16, const _Float16* __restrict__ w16,
    const float* __restrict__ b_ih, const float* __restrict__ b_hh,
    float* __restrict__ gx) {
  const int tid  = threadIdx.x;
  const int lane = tid & 31;
  const int wv   = tid >> 5;   // 0..7
  const int mw   = wv >> 1;    // 0..3  (16-row strip)
  const int nw   = wv & 1;     // 0..1  (64-col strip)
  const int m0   = blockIdx.x * 64;
  const int n0   = blockIdx.y * 128;

  const int arow = m0 + 16 * mw + (lane & 15);
  const int kx8  = (lane >> 4) * 8;    // A-frag K offset
  const int kb16 = (lane >> 4) * 16;   // unused (B-frag also uses kx8)
  const _Float16* abase = x16 + (size_t)arow * DD + kx8;

  v8f acc[4];
#pragma unroll
  for (int i = 0; i < 4; ++i)
#pragma unroll
    for (int r = 0; r < 8; ++r) acc[i][r] = 0.0f;

  // prologue: fragments for k-chunk 0
  v16h a0 = ld_frag2(abase, abase + 16);
  v16h b0[4];
#pragma unroll
  for (int nt = 0; nt < 4; ++nt) {
    const int n = n0 + 64 * nw + 16 * nt + (lane & 15);
    const _Float16* bp = w16 + (size_t)n * DD + kx8;
    b0[nt] = ld_frag2(bp, bp + 16);
  }

#pragma unroll
  for (int kc = 0; kc < 8; ++kc) {   // K = 256 in chunks of 32
    // fetch next chunk's fragments (redundant reload of chunk 7 on last iter,
    // served from L0) before consuming the current ones.
    const int kn = (kc < 7) ? kc + 1 : 7;
    v16h a1 = ld_frag2(abase + kn * 32, abase + kn * 32 + 16);
    v16h b1[4];
#pragma unroll
    for (int nt = 0; nt < 4; ++nt) {
      const int n = n0 + 64 * nw + 16 * nt + (lane & 15);
      const _Float16* bp = w16 + (size_t)n * DD + kn * 32 + kx8;
      b1[nt] = ld_frag2(bp, bp + 16);
    }
#pragma unroll
    for (int nt = 0; nt < 4; ++nt)
      acc[nt] = __builtin_amdgcn_wmma_f32_16x16x32_f16(
          false, a0, false, b0[nt], (short)0, acc[nt], false, false);
    a0 = a1;
#pragma unroll
    for (int nt = 0; nt < 4; ++nt) b0[nt] = b1[nt];
  }

  // epilogue: fused bias (N uniform across the 8 accumulator VGPRs),
  // store f32 into time-major gx[t][b][n]
  const int btRow = m0 + 16 * mw + 8 * (lane >> 4);
#pragma unroll
  for (int nt = 0; nt < 4; ++nt) {
    const int n  = n0 + 64 * nw + 16 * nt + (lane & 15);
    const float bv = b_ih[n] + b_hh[n];
#pragma unroll
    for (int r = 0; r < 8; ++r) {
      const int bt = btRow + r;
      const int bb = bt >> 10;        // bt / TT
      const int t  = bt & (TT - 1);   // bt % TT
      gx[((size_t)t * BB + bb) * NG + n] = acc[nt][r] + bv;
    }
  }
}

// ---------------------------------------------------------------------------
// LSTM recurrence scan. grid = (4 batch-tiles, 2 directions), block = 256 (8 waves).
// Wave w owns h cols [16w,16w+16) and its 4 gate tiles; W_hh B-fragments live
// in 128 VGPRs (loaded once from the f16 copy), h double-buffered in 8KB LDS.
// gx is added AFTER the WMMA chain so the chain's critical path never waits on
// global memory -- gx loads overlap the 16 WMMAs.
__global__ __launch_bounds__(256, 1) void lstm_scan_kernel(
    const float* __restrict__ gxF, const float* __restrict__ gxR,   // [T,B,512]
    const _Float16* __restrict__ whhF16, const _Float16* __restrict__ whhR16, // [512,128]
    float* __restrict__ out /* [B,T,256] */) {
  const int tid   = threadIdx.x;
  const int lane  = tid & 31;
  const int wv    = tid >> 5;          // 0..7 -> h column tile
  const int mtile = blockIdx.x;        // 0..3 -> batch rows 16*mtile
  const bool rev  = (blockIdx.y != 0);
  const float*    gx    = rev ? gxR : gxF;
  const _Float16* whh16 = rev ? whhR16 : whhF16;
  const int colOff = rev ? HH : 0;

  __shared__ _Float16 h16[2][16][HH];

  const int nloc = lane & 15;
  const int kx8  = (lane >> 4) * 8;
  const int kb16 = (lane >> 4) * 16;

  // Preload W_hh^T B-fragments into VGPRs: bf[gate][kchunk] (128 VGPRs/lane).
  v16h bf[4][4];
#pragma unroll
  for (int g = 0; g < 4; ++g) {
    const int n = g * HH + wv * 16 + nloc;   // gate column (row of whh)
#pragma unroll
    for (int kc = 0; kc < 4; ++kc) {
      const _Float16* bp = whh16 + (size_t)n * HH + kc * 32 + kx8;
      bf[g][kc] = ld_frag2(bp, bp + 16);
    }
  }

  // zero the h buffer read by step 0 (buffer 1)
  for (int e = tid; e < 16 * HH; e += 256)
    (&h16[1][0][0])[e] = (_Float16)0.0f;
  __syncthreads();

  v8f c;
#pragma unroll
  for (int r = 0; r < 8; ++r) c[r] = 0.0f;

  const int mrowBase = 16 * mtile + 8 * (lane >> 4);   // global batch row base

  for (int s = 0; s < TT; ++s) {
    const int t  = rev ? (TT - 1 - s) : s;
    const int pb = (s + 1) & 1;      // buffer holding h_{t-1}
    const int cb = s & 1;            // buffer receiving h_t

    // Issue gx[t] loads first (independent of the WMMA chain; overlap fully).
    v8f gxv[4];
#pragma unroll
    for (int g = 0; g < 4; ++g) {
      const int n = g * HH + wv * 16 + nloc;
      const float* gp = gx + ((size_t)t * BB + mrowBase) * NG + n;
#pragma unroll
      for (int r = 0; r < 8; ++r)
        gxv[g][r] = gp[(size_t)r * NG];
    }
    // prefetch next step's gx tile (global_prefetch_b8)
    {
      const int t2 = rev ? (t > 0 ? t - 1 : 0) : (t < TT - 1 ? t + 1 : t);
#pragma unroll
      for (int g = 0; g < 4; ++g)
        __builtin_prefetch(gx + ((size_t)t2 * BB + mrowBase) * NG + g * HH + wv * 16 + nloc, 0, 1);
    }

    // A fragments from previous h (f16 in LDS): 2x ds_load_b128 each
    v16h af[4];
#pragma unroll
    for (int kc = 0; kc < 4; ++kc) {
      const _Float16* ap = &h16[pb][lane & 15][kc * 32 + kx8];
      af[kc] = ld_frag2(ap, ap + 16);
    }

    // 16 WMMAs from zero C: acc = h_{t-1} @ W_hh^T (4 independent chains)
    v8f acc[4];
#pragma unroll
    for (int g = 0; g < 4; ++g) {
      v8f z;
#pragma unroll
      for (int r = 0; r < 8; ++r) z[r] = 0.0f;
      acc[g] = z;
#pragma unroll
      for (int kc = 0; kc < 4; ++kc)
        acc[g] = __builtin_amdgcn_wmma_f32_16x16x32_f16(
            false, af[kc], false, bf[g][kc], (short)0, acc[g], false, false);
    }

    // LSTM cell (torch gate order i,f,g,o); gx added here, c in accumulator layout
#pragma unroll
    for (int r = 0; r < 8; ++r) {
      const float ig = sigmf(acc[0][r] + gxv[0][r]);
      const float fg = sigmf(acc[1][r] + gxv[1][r]);
      const float gg = tanhf(acc[2][r] + gxv[2][r]);
      const float og = sigmf(acc[3][r] + gxv[3][r]);
      const float cc = fg * c[r] + ig * gg;
      c[r] = cc;
      const float h = og * tanhf(cc);
      const int b = mrowBase + r;
      out[((size_t)b * TT + t) * (2 * HH) + colOff + wv * 16 + nloc] = h;
      h16[cb][8 * (lane >> 4) + r][wv * 16 + nloc] = (_Float16)h;
    }
    __syncthreads();
  }
}

// ---------------------------------------------------------------------------
// emissions = out1 @ fc_w^T + fc_b  (N=3, scalar tail)
__global__ void emis_kernel(const float* __restrict__ xin,
                            const float* __restrict__ fcw,
                            const float* __restrict__ fcb,
                            float* __restrict__ em) {
  const int bt = blockIdx.x * blockDim.x + threadIdx.x;
  if (bt >= BT) return;
  const float* xr = xin + (size_t)bt * (2 * HH);
  float s0 = fcb[0], s1 = fcb[1], s2 = fcb[2];
  for (int d = 0; d < 2 * HH; ++d) {
    const float xv = xr[d];
    s0 = fmaf(xv, fcw[d], s0);
    s1 = fmaf(xv, fcw[2 * HH + d], s1);
    s2 = fmaf(xv, fcw[4 * HH + d], s2);
  }
  em[(size_t)bt * 3 + 0] = s0;
  em[(size_t)bt * 3 + 1] = s1;
  em[(size_t)bt * 3 + 2] = s2;
}

// ---------------------------------------------------------------------------
// CRF NLL: one thread per batch element, sequential DP over T (NTAG=3).
__device__ __forceinline__ float lse3(float x, float y, float z) {
  const float m = fmaxf(x, fmaxf(y, z));
  return m + __logf(__expf(x - m) + __expf(y - m) + __expf(z - m));
}

__global__ __launch_bounds__(64) void crf_kernel(
    const float* __restrict__ em, const int* __restrict__ tags,
    const float* __restrict__ start, const float* __restrict__ endp,
    const float* __restrict__ trans, float* __restrict__ loss_out) {
  const int b = threadIdx.x;  // 64 threads, 1 block
  const float* e  = em + (size_t)b * TT * 3;
  const int*   tg = tags + (size_t)b * TT;
  float tr[9], st[3], en[3];
#pragma unroll
  for (int i = 0; i < 9; ++i) tr[i] = trans[i];
#pragma unroll
  for (int i = 0; i < 3; ++i) { st[i] = start[i]; en[i] = endp[i]; }

  int prev = tg[0];
  float num = st[prev] + e[prev];
  float a0 = st[0] + e[0], a1 = st[1] + e[1], a2 = st[2] + e[2];
  for (int t = 1; t < TT; ++t) {
    const int ct = tg[t];
    const float* et = e + (size_t)t * 3;
    num += et[ct] + tr[prev * 3 + ct];
    prev = ct;
    const float n0 = lse3(a0 + tr[0], a1 + tr[3], a2 + tr[6]) + et[0];
    const float n1 = lse3(a0 + tr[1], a1 + tr[4], a2 + tr[7]) + et[1];
    const float n2 = lse3(a0 + tr[2], a1 + tr[5], a2 + tr[8]) + et[2];
    a0 = n0; a1 = n1; a2 = n2;
  }
  num += en[prev];
  const float logZ = lse3(a0 + en[0], a1 + en[1], a2 + en[2]);

  __shared__ float red[64];
  red[b] = num - logZ;
  __syncthreads();
  if (b == 0) {
    float s = 0.0f;
    for (int i = 0; i < 64; ++i) s += red[i];
    loss_out[0] = -s / 64.0f;
  }
}

// ---------------------------------------------------------------------------
extern "C" void kernel_launch(void* const* d_in, const int* in_sizes, int n_in,
                              void* d_out, int out_size, void* d_ws, size_t ws_size,
                              hipStream_t stream) {
  (void)in_sizes; (void)n_in; (void)out_size; (void)ws_size;
  const float* x        = (const float*)d_in[0];
  const int*   tags     = (const int*)d_in[1];
  const float* w_ih_l0  = (const float*)d_in[2];
  const float* w_hh_l0  = (const float*)d_in[3];
  const float* b_ih_l0  = (const float*)d_in[4];
  const float* b_hh_l0  = (const float*)d_in[5];
  const float* w_ih_l0r = (const float*)d_in[6];
  const float* w_hh_l0r = (const float*)d_in[7];
  const float* b_ih_l0r = (const float*)d_in[8];
  const float* b_hh_l0r = (const float*)d_in[9];
  const float* w_ih_l1  = (const float*)d_in[10];
  const float* w_hh_l1  = (const float*)d_in[11];
  const float* b_ih_l1  = (const float*)d_in[12];
  const float* b_hh_l1  = (const float*)d_in[13];
  const float* w_ih_l1r = (const float*)d_in[14];
  const float* w_hh_l1r = (const float*)d_in[15];
  const float* b_ih_l1r = (const float*)d_in[16];
  const float* b_hh_l1r = (const float*)d_in[17];
  const float* fc_w     = (const float*)d_in[18];
  const float* fc_b     = (const float*)d_in[19];
  const float* crf_s    = (const float*)d_in[20];
  const float* crf_e    = (const float*)d_in[21];
  const float* crf_t    = (const float*)d_in[22];
  float* out = (float*)d_out;

  // workspace layout (all offsets 16B-aligned)
  char* ws = (char*)d_ws;
  size_t off = 0;
  _Float16* x16   = (_Float16*)(ws + off); off += (size_t)BT * DD * 2;   // 32 MB
  float*    gxF   = (float*)(ws + off);    off += (size_t)BT * NG * 4;   // 128 MB
  float*    gxR   = (float*)(ws + off);    off += (size_t)BT * NG * 4;   // 128 MB
  float*    out01 = (float*)(ws + off);    off += (size_t)BT * DD * 4;   // 64 MB (layer0 & layer1 output)
  _Float16* wF16  = (_Float16*)(ws + off); off += (size_t)NG * DD * 2;   // 256 KB
  _Float16* wR16  = (_Float16*)(ws + off); off += (size_t)NG * DD * 2;
  _Float16* whF16 = (_Float16*)(ws + off); off += (size_t)NG * HH * 2;   // 128 KB
  _Float16* whR16 = (_Float16*)(ws + off); off += (size_t)NG * HH * 2;

  const dim3 gemmGrid(BT / 64, NG / 128);
  const dim3 scanGrid(4, 2);
  const int NW_IH = NG * DD;   // 131072
  const int NW_HH = NG * HH;   // 65536

  // ---- layer 0 ----
  cvt_f16_kernel<<<4096, 256, 0, stream>>>(x, x16, BT * DD);
  cvt_f16_kernel<<<512, 256, 0, stream>>>(w_ih_l0,  wF16,  NW_IH);
  cvt_f16_kernel<<<512, 256, 0, stream>>>(w_ih_l0r, wR16,  NW_IH);
  cvt_f16_kernel<<<256, 256, 0, stream>>>(w_hh_l0,  whF16, NW_HH);
  cvt_f16_kernel<<<256, 256, 0, stream>>>(w_hh_l0r, whR16, NW_HH);
  gemm_gx_kernel<<<gemmGrid, 256, 0, stream>>>(x16, wF16, b_ih_l0,  b_hh_l0,  gxF);
  gemm_gx_kernel<<<gemmGrid, 256, 0, stream>>>(x16, wR16, b_ih_l0r, b_hh_l0r, gxR);
  lstm_scan_kernel<<<scanGrid, 256, 0, stream>>>(gxF, gxR, whF16, whR16, out01);

  // ---- layer 1 ----
  cvt_f16_kernel<<<4096, 256, 0, stream>>>(out01, x16, BT * DD);
  cvt_f16_kernel<<<512, 256, 0, stream>>>(w_ih_l1,  wF16,  NW_IH);
  cvt_f16_kernel<<<512, 256, 0, stream>>>(w_ih_l1r, wR16,  NW_IH);
  cvt_f16_kernel<<<256, 256, 0, stream>>>(w_hh_l1,  whF16, NW_HH);
  cvt_f16_kernel<<<256, 256, 0, stream>>>(w_hh_l1r, whR16, NW_HH);
  gemm_gx_kernel<<<gemmGrid, 256, 0, stream>>>(x16, wF16, b_ih_l1,  b_hh_l1,  gxF);
  gemm_gx_kernel<<<gemmGrid, 256, 0, stream>>>(x16, wR16, b_ih_l1r, b_hh_l1r, gxR);
  lstm_scan_kernel<<<scanGrid, 256, 0, stream>>>(gxF, gxR, whF16, whR16, out01);

  // ---- head: emissions (to d_out+1) then CRF loss (to d_out[0]) ----
  emis_kernel<<<BT / 256, 256, 0, stream>>>(out01, fc_w, fc_b, out + 1);
  crf_kernel<<<1, 64, 0, stream>>>(out + 1, tags, crf_s, crf_e, crf_t, out);
}
